// Block_63591285784865
// MI455X (gfx1250) — matrix operand order repaired and checked by reading, compiled, not run on hardware
//
#include <hip/hip_runtime.h>

// ---------------------------------------------------------------------------
// Transformer block (LN -> MHA -> residual -> LN -> dense 8-expert MoE) for
// MI455X / gfx1250.  All GEMMs run on v_wmma_f32_16x16x32_bf16 (wave32).
// GEMM: 128x128 block tile, 8 waves, pair-packed B in LDS, double-buffered
// K-loop (global loads overlap WMMA), K-tile prefetch (global_prefetch_b8).
// ---------------------------------------------------------------------------

namespace {
constexpr int B_  = 4;
constexpr int T_  = 1024;
constexpr int C_  = 512;
constexpr int H_  = 8;
constexpr int HD_ = 64;
constexpr int NE_ = 8;
constexpr int FF_ = 4 * C_;   // 2048
constexpr int BT_ = B_ * T_;  // 4096
}

typedef __bf16 bf16;
typedef __attribute__((ext_vector_type(16))) __bf16 v16bf;
typedef __attribute__((ext_vector_type(8)))  float  v8f;

union Frag {
  v16bf v;
  bf16 h[16];
  unsigned u[8];
};

__device__ __forceinline__ v8f v8zero() {
  v8f z = {0.f, 0.f, 0.f, 0.f, 0.f, 0.f, 0.f, 0.f};
  return z;
}

__device__ __forceinline__ v8f wmma_bf16(const Frag& a, const Frag& b, v8f c) {
  // D = A(16x32 bf16) * B(32x16 bf16) + C(16x16 f32)
  return __builtin_amdgcn_wmma_f32_16x16x32_bf16(
      /*neg_a=*/false, a.v, /*neg_b=*/false, b.v,
      /*c_mod=*/(short)0, c, /*reuse_a=*/false, /*reuse_b=*/false);
}

// A-fragment K index for VGPR v (contiguous order k = khalf*16 + 2v, same order as the B fragments)
__device__ __forceinline__ int a_kidx(int v, int khalf) {
  return (khalf << 4) + (v << 1);
}

// pack two rows of 8 bf16 into 8 pair-packed dwords
__device__ __forceinline__ void pack_pairs(const uint4& r0, const uint4& r1,
                                           uint4& o0, uint4& o1) {
  const unsigned short* e0 = (const unsigned short*)&r0;
  const unsigned short* e1 = (const unsigned short*)&r1;
  o0.x = e0[0] | ((unsigned)e1[0] << 16);
  o0.y = e0[1] | ((unsigned)e1[1] << 16);
  o0.z = e0[2] | ((unsigned)e1[2] << 16);
  o0.w = e0[3] | ((unsigned)e1[3] << 16);
  o1.x = e0[4] | ((unsigned)e1[4] << 16);
  o1.y = e0[5] | ((unsigned)e1[5] << 16);
  o1.z = e0[6] | ((unsigned)e1[6] << 16);
  o1.w = e0[7] | ((unsigned)e1[7] << 16);
}

// ---------------------------------------------------------------------------
// f32 -> bf16 conversion (weights)
// ---------------------------------------------------------------------------
__global__ void convert_kernel(const float* __restrict__ src,
                               bf16* __restrict__ dst, int n) {
  int i = blockIdx.x * 256 + threadIdx.x;
  if (i < n) dst[i] = (bf16)src[i];
}

// ---------------------------------------------------------------------------
// LayerNorm: one wave per token (wave32), C = 512 -> 16 elems/lane
// ---------------------------------------------------------------------------
__global__ __launch_bounds__(256)
void ln_kernel(const float* __restrict__ x, const float* __restrict__ g,
               const float* __restrict__ be, bf16* __restrict__ out) {
  const int wave = threadIdx.x >> 5;
  const int lane = threadIdx.x & 31;
  const int tok  = blockIdx.x * 8 + wave;
  const float* xr = x + (size_t)tok * C_;

  float vals[16];
  float s = 0.f;
#pragma unroll
  for (int i = 0; i < 16; ++i) {
    vals[i] = xr[lane + i * 32];
    s += vals[i];
  }
#pragma unroll
  for (int off = 16; off >= 1; off >>= 1) s += __shfl_xor(s, off, 32);
  const float mu = s * (1.0f / C_);

  float s2 = 0.f;
#pragma unroll
  for (int i = 0; i < 16; ++i) {
    const float d = vals[i] - mu;
    s2 += d * d;
  }
#pragma unroll
  for (int off = 16; off >= 1; off >>= 1) s2 += __shfl_xor(s2, off, 32);
  const float inv = rsqrtf(s2 * (1.0f / C_) + 1e-5f);

#pragma unroll
  for (int i = 0; i < 16; ++i) {
    const int c = lane + i * 32;
    out[(size_t)tok * C_ + c] = (bf16)((vals[i] - mu) * inv * g[c] + be[c]);
  }
}

// ---------------------------------------------------------------------------
// Router: one wave per token; exact top-2 scatter-mask softmax semantics
// ---------------------------------------------------------------------------
__global__ __launch_bounds__(256)
void router_kernel(const bf16* __restrict__ h2, const float* __restrict__ Wr,
                   const float* __restrict__ br, float* __restrict__ router) {
  const int wave = threadIdx.x >> 5;
  const int lane = threadIdx.x & 31;
  const int tok  = blockIdx.x * 8 + wave;

  float acc[NE_];
#pragma unroll
  for (int e = 0; e < NE_; ++e) acc[e] = 0.f;

#pragma unroll
  for (int i = 0; i < 16; ++i) {
    const int c = lane + i * 32;
    const float hv = (float)h2[(size_t)tok * C_ + c];
#pragma unroll
    for (int e = 0; e < NE_; ++e) acc[e] += hv * Wr[c * NE_ + e];
  }
#pragma unroll
  for (int e = 0; e < NE_; ++e) {
#pragma unroll
    for (int off = 16; off >= 1; off >>= 1) acc[e] += __shfl_xor(acc[e], off, 32);
    acc[e] += br[e];
  }

  // top-2 (first occurrence wins ties, matching jax.lax.top_k)
  int i1 = 0;
#pragma unroll
  for (int e = 1; e < NE_; ++e) if (acc[e] > acc[i1]) i1 = e;
  int i2 = (i1 == 0) ? 1 : 0;
#pragma unroll
  for (int e = 0; e < NE_; ++e) if (e != i1 && acc[e] > acc[i2]) i2 = e;

  float mask[NE_];
#pragma unroll
  for (int e = 0; e < NE_; ++e)
    mask[e] = (e == i1) ? acc[i1] : ((e == i2) ? acc[i2] : 0.0f);
  float mx = mask[0];
#pragma unroll
  for (int e = 1; e < NE_; ++e) mx = fmaxf(mx, mask[e]);
  float ex[NE_], den = 0.f;
#pragma unroll
  for (int e = 0; e < NE_; ++e) { ex[e] = __expf(mask[e] - mx); den += ex[e]; }
  const float rden = 1.0f / den;
  if (lane == 0) {
#pragma unroll
    for (int e = 0; e < NE_; ++e) router[(size_t)tok * NE_ + e] = ex[e] * rden;
  }
}

// ---------------------------------------------------------------------------
// Generic bf16 WMMA GEMM.
//   Block tile 128x128, 8 waves (256 thr), K step 32, double-buffered LDS.
//   Wave w: rows (w&3)*32 .. +31 (2 m-subtiles), cols (w>>2)*64 .. +63
//   (4 n-subtiles) -> 8 WMMAs per K-step, B-frag shared across m-subtiles.
//   B staged pair-packed: sBp[kpair][n] = (B[2k][n], B[2k+1][n]) so a B
//   fragment is 8 dword LDS reads (contiguous K = khalf*16 + 2v, same order as A).
//   A row-major [M,lda]; B row-major [K,ldb] or head-blocked [H,C,HD].
// ---------------------------------------------------------------------------
enum { EPI_BF16 = 0, EPI_RELU_BF16 = 1, EPI_RESID_F32 = 2, EPI_MOE_ACC = 3 };

template <int EPI, bool HEADED>
__global__ __launch_bounds__(256)
void gemm_kernel(const bf16* __restrict__ A, int lda,
                 const bf16* __restrict__ B, int ldb, int K,
                 const float* __restrict__ bias,
                 const float* __restrict__ extra,   // residual [M,ldc] or router [M,NE]
                 float* __restrict__ outf, bf16* __restrict__ outb,
                 int ldc, int expert) {
  __shared__ __align__(16) bf16     sA[2][128 * 32];   // 2 x 8 KB
  __shared__ __align__(16) unsigned sBp[2][16 * 128];  // 2 x 8 KB, pair-packed

  const int tid   = threadIdx.x;
  const int wave  = tid >> 5;
  const int lane  = tid & 31;
  const int mlane = lane & 15;
  const int khalf = lane >> 4;
  const int mw    = (wave & 3) * 32;   // wave row offset in block
  const int nw    = (wave >> 2) * 64;  // wave col offset in block
  const int n0 = blockIdx.x * 128;
  const int m0 = blockIdx.y * 128;

  v8f acc[2][4];
#pragma unroll
  for (int ms = 0; ms < 2; ++ms)
#pragma unroll
    for (int nt = 0; nt < 4; ++nt) acc[ms][nt] = v8zero();

  // staging geometry
  const int ia = tid * 16;            // A: 128x32 bf16, 16 elems / thread
  const int ar = ia >> 5, ac = ia & 31;
  const int bt = tid * 8;             // B: 16x128 dwords, 8 dwords / thread
  const int kp = bt >> 7;             // K-pair index 0..15
  const int nb = bt & 127;            // col offset (multiple of 8)

  const int brstride = HEADED ? 64 : ldb;
  auto aptr = [&](int k0) -> const bf16* {
    return A + (size_t)(m0 + ar) * lda + k0 + ac;
  };
  auto bptr = [&](int k0) -> const bf16* {
    if (HEADED) {
      // B[k][n] = W[h][k][d], h = n>>6, d = n&63 ; nb is 8-aligned so the
      // 8-wide run never crosses a head boundary.
      const int gn = n0 + nb;
      return B + (size_t)(gn >> 6) * K * 64 + (size_t)(k0 + 2 * kp) * 64 + (gn & 63);
    }
    return B + (size_t)(k0 + 2 * kp) * ldb + n0 + nb;
  };

  const int nIter = K >> 5;

  // ---- preload tile 0 ----
  {
    const uint4* s4 = (const uint4*)aptr(0);
    const uint4 a0 = s4[0], a1 = s4[1];
    uint4* d4 = (uint4*)(sA[0] + ia);
    d4[0] = a0;
    d4[1] = a1;
    const bf16* bs = bptr(0);
    const uint4 r0 = *(const uint4*)bs;
    const uint4 r1 = *(const uint4*)(bs + brstride);
    uint4 o0, o1;
    pack_pairs(r0, r1, o0, o1);
    uint4* e4 = (uint4*)(sBp[0] + kp * 128 + nb);
    e4[0] = o0;
    e4[1] = o1;
  }
  __syncthreads();

  for (int it = 0; it < nIter; ++it) {
    const int cur = it & 1;
    const bool more = (it + 1) < nIter;

    // ---- issue next K-tile's global loads before compute (overlap) ----
    uint4 na0, na1, nr0, nr1;
    if (more) {
      const int k1 = (it + 1) << 5;
      const uint4* s4 = (const uint4*)aptr(k1);
      na0 = s4[0];
      na1 = s4[1];
      const bf16* bs = bptr(k1);
      nr0 = *(const uint4*)bs;
      nr1 = *(const uint4*)(bs + brstride);
      if (it + 2 < nIter) {
        const int k2 = (it + 2) << 5;
        __builtin_prefetch(aptr(k2), 0, 1);    // global_prefetch_b8
        __builtin_prefetch(bptr(k2), 0, 1);
      }
    }

    // ---- fragments + 8 WMMAs on current buffer ----
    Frag af[2];
#pragma unroll
    for (int ms = 0; ms < 2; ++ms) {
      const int arow = mw + ms * 16 + mlane;
#pragma unroll
      for (int v = 0; v < 8; ++v)
        af[ms].u[v] = *(const unsigned*)(sA[cur] + arow * 32 + a_kidx(v, khalf));
    }
#pragma unroll
    for (int nt = 0; nt < 4; ++nt) {
      Frag bfg;
      const int col = nw + nt * 16 + mlane;
#pragma unroll
      for (int v = 0; v < 8; ++v)
        bfg.u[v] = sBp[cur][(khalf * 8 + v) * 128 + col];
#pragma unroll
      for (int ms = 0; ms < 2; ++ms)
        acc[ms][nt] = wmma_bf16(af[ms], bfg, acc[ms][nt]);
    }

    // ---- stage next tile into the other buffer ----
    if (more) {
      const int nxt = cur ^ 1;
      uint4* d4 = (uint4*)(sA[nxt] + ia);
      d4[0] = na0;
      d4[1] = na1;
      uint4 o0, o1;
      pack_pairs(nr0, nr1, o0, o1);
      uint4* e4 = (uint4*)(sBp[nxt] + kp * 128 + nb);
      e4[0] = o0;
      e4[1] = o1;
    }
    __syncthreads();
  }

  // ---- epilogue ----
#pragma unroll
  for (int ms = 0; ms < 2; ++ms) {
#pragma unroll
    for (int nt = 0; nt < 4; ++nt) {
#pragma unroll
      for (int v = 0; v < 8; ++v) {
        const int row = m0 + mw + ms * 16 + v + (khalf << 3);
        const int col = n0 + nw + nt * 16 + mlane;
        const size_t oidx = (size_t)row * ldc + col;
        float val = acc[ms][nt][v];
        if (EPI == EPI_BF16) {
          outb[oidx] = (bf16)val;
        } else if (EPI == EPI_RELU_BF16) {
          val += bias[col];
          outb[oidx] = (bf16)fmaxf(val, 0.0f);
        } else if (EPI == EPI_RESID_F32) {
          outf[oidx] = extra[oidx] + val + bias[col];
        } else {  // EPI_MOE_ACC
          const float w = extra[(size_t)row * NE_ + expert];
          outf[oidx] += w * (val + bias[col]);
        }
      }
    }
  }
}

// ---------------------------------------------------------------------------
// Flash attention: 1 wave per 16 queries per (b,h); online softmax; WMMA for
// both Q*K^T and P*V.  Q/K/V stored bf16 as [B*T, H*HD] (ld = 512).
// ---------------------------------------------------------------------------
__global__ __launch_bounds__(32)
void attn_kernel(const bf16* __restrict__ Q, const bf16* __restrict__ Km,
                 const bf16* __restrict__ V, bf16* __restrict__ O) {
  const int qt = blockIdx.x;
  const int bh = blockIdx.y;
  const int b = bh >> 3, h = bh & 7;
  const int q0 = qt * 16;
  const int lane  = threadIdx.x;
  const int mlane = lane & 15;
  const int khalf = lane >> 4;

  const size_t rowbase = (size_t)b * T_ * C_;
  const int hcol = h * HD_;
  const bf16* Qb = Q + rowbase + hcol;
  const bf16* Kb = Km + rowbase + hcol;
  const bf16* Vb = V + rowbase + hcol;

  // Q A-fragments for both K-steps of HD=64 (constant across the key loop)
  Frag qa[2];
#pragma unroll
  for (int kk = 0; kk < 2; ++kk) {
#pragma unroll
    for (int v = 0; v < 8; ++v) {
      const int kd = kk * 32 + a_kidx(v, khalf);
      qa[kk].u[v] = *(const unsigned*)(Qb + (size_t)(q0 + mlane) * C_ + kd);
    }
  }

  v8f o[4];
#pragma unroll
  for (int nt = 0; nt < 4; ++nt) o[nt] = v8zero();
  float mst[8], lst[8];
#pragma unroll
  for (int v = 0; v < 8; ++v) { mst[v] = -1e30f; lst[v] = 0.f; }

  __shared__ float Pt[16][32];

  for (int s0 = 0; s0 <= q0; s0 += 32) {
    // S = Q * K^T for 16 queries x 32 keys (two 16x16 tiles)
    v8f sc[2];
    sc[0] = v8zero();
    sc[1] = v8zero();
#pragma unroll
    for (int st = 0; st < 2; ++st) {
#pragma unroll
      for (int kk = 0; kk < 2; ++kk) {
        Frag kf;
#pragma unroll
        for (int v = 0; v < 8; ++v) {
          const int kd = kk * 32 + (khalf << 4) + (v << 1);
          const int srow = s0 + st * 16 + mlane;
          kf.u[v] = *(const unsigned*)(Kb + (size_t)srow * C_ + kd);
        }
        sc[st] = wmma_bf16(qa[kk], kf, sc[st]);
      }
    }

    // causal mask + online softmax (rows live along VGPR index; 16-lane reduce)
    float scale8[8];
#pragma unroll
    for (int v = 0; v < 8; ++v) {
      const int mrow = q0 + v + (khalf << 3);
      float a0 = sc[0][v] * 0.125f;  // 1/sqrt(64)
      float a1 = sc[1][v] * 0.125f;
      if (s0 + mlane > mrow) a0 = -1e30f;
      if (s0 + 16 + mlane > mrow) a1 = -1e30f;
      float mx = fmaxf(a0, a1);
      mx = fmaxf(mx, __shfl_xor(mx, 1, 32));
      mx = fmaxf(mx, __shfl_xor(mx, 2, 32));
      mx = fmaxf(mx, __shfl_xor(mx, 4, 32));
      mx = fmaxf(mx, __shfl_xor(mx, 8, 32));
      const float nm = fmaxf(mst[v], mx);
      const float sf = __expf(mst[v] - nm);
      mst[v] = nm;
      const float p0 = __expf(a0 - nm);
      const float p1 = __expf(a1 - nm);
      sc[0][v] = p0;
      sc[1][v] = p1;
      float rs = p0 + p1;
      rs += __shfl_xor(rs, 1, 32);
      rs += __shfl_xor(rs, 2, 32);
      rs += __shfl_xor(rs, 4, 32);
      rs += __shfl_xor(rs, 8, 32);
      lst[v] = lst[v] * sf + rs;
      scale8[v] = sf;
    }
#pragma unroll
    for (int nt = 0; nt < 4; ++nt)
#pragma unroll
      for (int v = 0; v < 8; ++v) o[nt][v] *= scale8[v];

    // transpose P (C-layout -> A-layout) through LDS, pack to bf16
    __syncthreads();
#pragma unroll
    for (int v = 0; v < 8; ++v) {
      Pt[v + (khalf << 3)][mlane] = sc[0][v];
      Pt[v + (khalf << 3)][16 + mlane] = sc[1][v];
    }
    __syncthreads();
    Frag pa;
#pragma unroll
    for (int v = 0; v < 8; ++v) {
      const int kd = a_kidx(v, khalf);
      pa.h[2 * v]     = (bf16)Pt[mlane][kd];
      pa.h[2 * v + 1] = (bf16)Pt[mlane][kd + 1];
    }

    // O += P * V
#pragma unroll
    for (int nt = 0; nt < 4; ++nt) {
      Frag vf;
#pragma unroll
      for (int v = 0; v < 8; ++v) {
        const int srow = s0 + (khalf << 4) + (v << 1);
        const bf16* p = Vb + (size_t)srow * C_ + nt * 16 + mlane;
        const unsigned lo = *(const unsigned short*)p;
        const unsigned hi = *(const unsigned short*)(p + C_);
        vf.u[v] = lo | (hi << 16);
      }
      o[nt] = wmma_bf16(pa, vf, o[nt]);
    }
  }

  // normalize and store attn output bf16 at [b*T + row, h*64 + n]
#pragma unroll
  for (int nt = 0; nt < 4; ++nt) {
#pragma unroll
    for (int v = 0; v < 8; ++v) {
      const int row = q0 + v + (khalf << 3);
      const float val = o[nt][v] / lst[v];
      O[rowbase + (size_t)row * C_ + hcol + nt * 16 + mlane] = (bf16)val;
    }
  }
}

// ---------------------------------------------------------------------------
// Host orchestration
// ---------------------------------------------------------------------------
extern "C" void kernel_launch(void* const* d_in, const int* in_sizes, int n_in,
                              void* d_out, int out_size, void* d_ws, size_t ws_size,
                              hipStream_t stream) {
  (void)in_sizes; (void)n_in; (void)out_size; (void)ws_size;
  const float* x    = (const float*)d_in[0];
  const float* ln1g = (const float*)d_in[1];
  const float* ln1b = (const float*)d_in[2];
  const float* ln2g = (const float*)d_in[3];
  const float* ln2b = (const float*)d_in[4];
  const float* Wq   = (const float*)d_in[5];
  const float* Wk   = (const float*)d_in[6];
  const float* Wv   = (const float*)d_in[7];
  const float* Wp   = (const float*)d_in[8];
  const float* bp   = (const float*)d_in[9];
  const float* Wr   = (const float*)d_in[10];
  const float* br   = (const float*)d_in[11];
  const float* W1   = (const float*)d_in[12];
  const float* b1   = (const float*)d_in[13];
  const float* W2   = (const float*)d_in[14];
  const float* b2   = (const float*)d_in[15];
  float* out = (float*)d_out;

  char* ws = (char*)d_ws;
  size_t off = 0;
  auto alloc = [&](size_t bytes) -> char* {
    char* p = ws + off;
    off += (bytes + 255) & ~(size_t)255;
    return p;
  };

  bf16* Wq_b  = (bf16*)alloc((size_t)H_ * C_ * HD_ * 2);
  bf16* Wk_b  = (bf16*)alloc((size_t)H_ * C_ * HD_ * 2);
  bf16* Wv_b  = (bf16*)alloc((size_t)H_ * C_ * HD_ * 2);
  bf16* Wp_b  = (bf16*)alloc((size_t)C_ * C_ * 2);
  bf16* W1_b  = (bf16*)alloc((size_t)NE_ * C_ * FF_ * 2);
  bf16* W2_b  = (bf16*)alloc((size_t)NE_ * FF_ * C_ * 2);
  bf16* h1b   = (bf16*)alloc((size_t)BT_ * C_ * 2);
  bf16* qb    = (bf16*)alloc((size_t)BT_ * C_ * 2);
  bf16* kb    = (bf16*)alloc((size_t)BT_ * C_ * 2);
  bf16* vb    = (bf16*)alloc((size_t)BT_ * C_ * 2);
  bf16* attnb = (bf16*)alloc((size_t)BT_ * C_ * 2);
  bf16* h2b   = (bf16*)alloc((size_t)BT_ * C_ * 2);
  bf16* hid   = (bf16*)alloc((size_t)BT_ * FF_ * 2);
  float* routerw = (float*)alloc((size_t)BT_ * NE_ * 4);

  auto cvt = [&](const float* s, bf16* d, int n) {
    convert_kernel<<<dim3((n + 255) / 256), 256, 0, stream>>>(s, d, n);
  };
  cvt(Wq, Wq_b, H_ * C_ * HD_);
  cvt(Wk, Wk_b, H_ * C_ * HD_);
  cvt(Wv, Wv_b, H_ * C_ * HD_);
  cvt(Wp, Wp_b, C_ * C_);
  cvt(W1, W1_b, NE_ * C_ * FF_);
  cvt(W2, W2_b, NE_ * FF_ * C_);

  // h = LN1(x)
  ln_kernel<<<dim3(BT_ / 8), 256, 0, stream>>>(x, ln1g, ln1b, h1b);

  // q,k,v = h @ W{q,k,v} (head-blocked weights), stored [B*T, H*HD] bf16
  const dim3 gQKV(C_ / 128, BT_ / 128);
  gemm_kernel<EPI_BF16, true><<<gQKV, 256, 0, stream>>>(
      h1b, C_, Wq_b, 0, C_, nullptr, nullptr, nullptr, qb, C_, 0);
  gemm_kernel<EPI_BF16, true><<<gQKV, 256, 0, stream>>>(
      h1b, C_, Wk_b, 0, C_, nullptr, nullptr, nullptr, kb, C_, 0);
  gemm_kernel<EPI_BF16, true><<<gQKV, 256, 0, stream>>>(
      h1b, C_, Wv_b, 0, C_, nullptr, nullptr, nullptr, vb, C_, 0);

  // causal flash attention per (b,h)
  attn_kernel<<<dim3(T_ / 16, B_ * H_), 32, 0, stream>>>(qb, kb, vb, attnb);

  // x2 = x + attn @ Wp + bp  -> d_out (f32)
  gemm_kernel<EPI_RESID_F32, false><<<dim3(C_ / 128, BT_ / 128), 256, 0, stream>>>(
      attnb, C_, Wp_b, C_, C_, bp, x, out, nullptr, C_, 0);

  // h2 = LN2(x2)
  ln_kernel<<<dim3(BT_ / 8), 256, 0, stream>>>(out, ln2g, ln2b, h2b);

  // router weights (top-2 scatter-mask softmax)
  router_kernel<<<dim3(BT_ / 8), 256, 0, stream>>>(h2b, Wr, br, routerw);

  // dense MoE: all experts, router-weighted accumulate into d_out
  for (int e = 0; e < NE_; ++e) {
    gemm_kernel<EPI_RELU_BF16, false><<<dim3(FF_ / 128, BT_ / 128), 256, 0, stream>>>(
        h2b, C_, W1_b + (size_t)e * C_ * FF_, FF_, C_,
        b1 + (size_t)e * FF_, nullptr, nullptr, hid, FF_, 0);
    gemm_kernel<EPI_MOE_ACC, false><<<dim3(C_ / 128, BT_ / 128), 256, 0, stream>>>(
        hid, FF_, W2_b + (size_t)e * FF_ * C_, C_, FF_,
        b2 + (size_t)e * C_, routerw, out, nullptr, C_, e);
  }
}
